// GlobalAttention_79233556677225
// MI455X (gfx1250) — matrix operand order repaired and checked by reading, not compiled, not run on hardware
//
#include <hip/hip_runtime.h>
#include <hip/hip_bf16.h>
#include <math.h>

// ---------------------------------------------------------------------------
// GlobalAttention on MI455X (gfx1250, wave32, WMMA).
// All GEMMs on v_wmma_f32_16x16x32_bf16 with bf16x3 split precision.
// f32->bf16 hi/lo splits are hoisted into a one-shot prep pass so the
// GEMM inner loops are pure {b128 loads + WMMA}.
// ---------------------------------------------------------------------------

typedef __attribute__((ext_vector_type(16))) __bf16 v16bf;
typedef __attribute__((ext_vector_type(8)))  __bf16 v8bf;
typedef __attribute__((ext_vector_type(8)))  float  v8f;

#define B_DIM 1024   // batch
#define HID   512    // HIDRY / DIM_OUT
#define CTXD  1024   // CTX
#define NS    13     // selected memory slices
#define KCAT  2560   // HID + 2*CTX

// mem[:, s, :] = memory_bank[src_idx(s), :, :]
__device__ __forceinline__ int src_idx(int s) {
  return (s < 6) ? (162 + s) : (23 + 24 * (s - 6));
}

// --- bf16x3 WMMA: acc += (ahi+alo) x (bhi+blo), dropping alo*blo -----------
__device__ __forceinline__ v8f wmma_bf16x3(const v16bf& ahi, const v16bf& alo,
                                           const v16bf& bhi, const v16bf& blo,
                                           v8f acc) {
  acc = __builtin_amdgcn_wmma_f32_16x16x32_bf16(false, alo, false, bhi,
                                                (short)0, acc, false, false);
  acc = __builtin_amdgcn_wmma_f32_16x16x32_bf16(false, ahi, false, blo,
                                                (short)0, acc, false, false);
  acc = __builtin_amdgcn_wmma_f32_16x16x32_bf16(false, ahi, false, bhi,
                                                (short)0, acc, false, false);
  return acc;
}

__device__ __forceinline__ v16bf cat16(v8bf a, v8bf b) {
  return __builtin_shufflevector(a, b, 0, 1, 2, 3, 4, 5, 6, 7,
                                 8, 9, 10, 11, 12, 13, 14, 15);
}

// A fragment (16x32 MxK) from pre-split bf16 row (ISA 7.12.2):
//   elems 0..7  -> K = kbase + (lane>>4)*8 + j
//   elems 8..15 -> K = kbase + 16 + (lane>>4)*8 + (j-8)
// Row base and kbase are 8-element aligned -> two b128 loads each.
__device__ __forceinline__ void load_a_frag_bf(const __bf16* __restrict__ rhi,
                                               const __bf16* __restrict__ rlo,
                                               int kbase, int lane,
                                               v16bf& hi, v16bf& lo) {
  int kg = (lane >> 4) * 8;
  hi = cat16(*(const v8bf*)(rhi + kbase + kg),
             *(const v8bf*)(rhi + kbase + 16 + kg));
  lo = cat16(*(const v8bf*)(rlo + kbase + kg),
             *(const v8bf*)(rlo + kbase + 16 + kg));
}

// B fragment (32x16 KxN) for out[m][n] = sum_k A[m][k]*W[n][k]:
//   lane holds N = lane&15, elems 0..7 -> K = kbase + (lane>>4)*8 + j,
//   elems 8..15 -> K = kbase + 16 + (lane>>4)*8 + (j-8)
// two 8-element runs -> two b128 loads per plane.
__device__ __forceinline__ void load_b_frag_bf(const __bf16* __restrict__ Whi,
                                               const __bf16* __restrict__ Wlo,
                                               int ldw, int nbase, int kbase, int lane,
                                               v16bf& hi, v16bf& lo) {
  size_t off = (size_t)(nbase + (lane & 15)) * ldw + kbase + (lane >> 4) * 8;
  hi = cat16(*(const v8bf*)(Whi + off), *(const v8bf*)(Whi + off + 16));
  lo = cat16(*(const v8bf*)(Wlo + off), *(const v8bf*)(Wlo + off + 16));
}

// ---------------------------------------------------------------------------
// Prep: split f32 array into bf16 hi/lo planes (one-shot, memory bound)
// ---------------------------------------------------------------------------
__global__ __launch_bounds__(256) void k_split(const float* __restrict__ src,
                                               __bf16* __restrict__ hi,
                                               __bf16* __restrict__ lo, int n) {
  int i = blockIdx.x * 256 + threadIdx.x;
  int stride = gridDim.x * 256;
  for (; i < n; i += stride) {
    float f = src[i];
    __bf16 h = (__bf16)f;
    hi[i] = h;
    lo[i] = (__bf16)(f - (float)h);
  }
}

// ---------------------------------------------------------------------------
// Kernel A: wq[b][o] = x[b]·W_q[o] + b_q[o]   (1024x512, K=512)
// ---------------------------------------------------------------------------
__global__ __launch_bounds__(256) void k_wq(const __bf16* __restrict__ xhi,
                                            const __bf16* __restrict__ xlo,
                                            const __bf16* __restrict__ Whi,
                                            const __bf16* __restrict__ Wlo,
                                            const float* __restrict__ bq,
                                            float* __restrict__ wq_out) {
  int lane = threadIdx.x & 31;
  int wave = threadIdx.x >> 5;
  int tile = blockIdx.x * 8 + wave;       // 0..2047
  int mbase = (tile >> 5) * 16;
  int nbase = (tile & 31) * 16;
  const __bf16* rhi = xhi + (size_t)(mbase + (lane & 15)) * HID;
  const __bf16* rlo = xlo + (size_t)(mbase + (lane & 15)) * HID;
  v8f acc = {};
  for (int k = 0; k < HID; k += 32) {
    v16bf ahi, alo, bhi, blo;
    load_a_frag_bf(rhi, rlo, k, lane, ahi, alo);
    load_b_frag_bf(Whi, Wlo, HID, nbase, k, lane, bhi, blo);
    acc = wmma_bf16x3(ahi, alo, bhi, blo, acc);
  }
  int col = nbase + (lane & 15);
  float bias = bq[col];
#pragma unroll
  for (int r = 0; r < 8; ++r) {
    int row = mbase + r + 8 * (lane >> 4);
    wq_out[(size_t)row * HID + col] = acc[r] + bias;
  }
}

// ---------------------------------------------------------------------------
// Kernel B (dominant, fused): 16 gathered rows (b,s) x all 512 cols, K=1024.
// A rows split into LDS (once per element); W_ctx pre-split in ws.
// scores[b,s] = sum_h tanh(wq[b,h] + uh) * v[h]; uh never touches HBM.
// ---------------------------------------------------------------------------
__global__ __launch_bounds__(256) void k_uh_scores(const float* __restrict__ mb,
                                                   const __bf16* __restrict__ Whi,
                                                   const __bf16* __restrict__ Wlo,
                                                   const float* __restrict__ wq,
                                                   const float* __restrict__ vvec,
                                                   float* __restrict__ scores) {
  __shared__ __attribute__((aligned(16))) __bf16 sAhi[16 * 512];
  __shared__ __attribute__((aligned(16))) __bf16 sAlo[16 * 512];
  __shared__ float sAcc[16];

  int tid  = threadIdx.x;
  int lane = tid & 31;
  int wave = tid >> 5;
  int rb   = blockIdx.x * 16;             // first gathered row of this block
  if (tid < 16) sAcc[tid] = 0.0f;

  v8f acc[4];
#pragma unroll
  for (int t = 0; t < 4; ++t) acc[t] = {};

  for (int ko = 0; ko < CTXD; ko += 512) {
    __syncthreads();                      // protect LDS reuse + sAcc init
    // stage A[16][512] (gathered, f32 -> bf16 hi/lo, once per element)
    for (int e = tid; e < 16 * 512; e += 256) {
      int r = e >> 9;
      int k = e & 511;
      int row = rb + r;
      int b = row / NS;
      int s = row - b * NS;
      float f = mb[((size_t)src_idx(s) * B_DIM + b) * CTXD + ko + k];
      __bf16 h = (__bf16)f;
      sAhi[e] = h;
      sAlo[e] = (__bf16)(f - (float)h);
    }
    __syncthreads();

#pragma unroll
    for (int t = 0; t < 4; ++t) {         // each wave owns 4 column tiles
      int nbase = (wave * 4 + t) * 16;
      for (int kb = 0; kb < 512; kb += 32) {
        v16bf ahi, alo, bhi, blo;
        int kg = (lane >> 4) * 8;
        int ab = (lane & 15) * 512 + kb + kg;   // 16B aligned runs
        ahi = cat16(*(const v8bf*)(sAhi + ab), *(const v8bf*)(sAhi + ab + 16));
        alo = cat16(*(const v8bf*)(sAlo + ab), *(const v8bf*)(sAlo + ab + 16));
        load_b_frag_bf(Whi, Wlo, CTXD, nbase, ko + kb, lane, bhi, blo);
        acc[t] = wmma_bf16x3(ahi, alo, bhi, blo, acc[t]);
      }
    }
  }

  // fused score reduction: tanh(wq + uh) * v summed over h
  float sc[8];
#pragma unroll
  for (int r = 0; r < 8; ++r) sc[r] = 0.0f;
#pragma unroll
  for (int t = 0; t < 4; ++t) {
    int col = (wave * 4 + t) * 16 + (lane & 15);
    float vn = vvec[col];
#pragma unroll
    for (int r = 0; r < 8; ++r) {
      int row = rb + r + 8 * (lane >> 4);
      int b = row / NS;
      float w = wq[(size_t)b * HID + col];
      sc[r] += tanhf(w + acc[t][r]) * vn;
    }
  }
#pragma unroll
  for (int m = 1; m <= 8; m <<= 1) {      // reduce across 16 cols per half
#pragma unroll
    for (int r = 0; r < 8; ++r) sc[r] += __shfl_xor(sc[r], m, 32);
  }
  if ((lane & 15) == 0) {
#pragma unroll
    for (int r = 0; r < 8; ++r)
      atomicAdd(&sAcc[r + 8 * (lane >> 4)], sc[r]);   // ds_add_f32
  }
  __syncthreads();
  if (tid < 16) scores[(size_t)rb + tid] = sAcc[tid];
}

// ---------------------------------------------------------------------------
// Kernel C: per-batch softmax(13) -> align output; context c = align·mem,
// emitted directly as bf16 hi/lo planes for kernel D.  One wave per batch.
// ---------------------------------------------------------------------------
__global__ __launch_bounds__(256) void k_softmax_ctx(const float* __restrict__ mb,
                                                     const float* __restrict__ scores,
                                                     float* __restrict__ align_out,
                                                     __bf16* __restrict__ chi,
                                                     __bf16* __restrict__ clo) {
  int lane = threadIdx.x & 31;
  int wave = threadIdx.x >> 5;
  int b = blockIdx.x * 8 + wave;          // 0..1023
  float sc = (lane < NS) ? scores[b * NS + lane] : -3.0e38f;
  float mx = sc;
#pragma unroll
  for (int m = 16; m >= 1; m >>= 1) mx = fmaxf(mx, __shfl_xor(mx, m, 32));
  float e = (lane < NS) ? __expf(sc - mx) : 0.0f;
  float sum = e;
#pragma unroll
  for (int m = 16; m >= 1; m >>= 1) sum += __shfl_xor(sum, m, 32);
  float al = e / sum;
  if (lane < NS) align_out[b * NS + lane] = al;

  float av[NS];
#pragma unroll
  for (int s = 0; s < NS; ++s) av[s] = __shfl(al, s, 32);

  for (int col = lane; col < CTXD; col += 32) {
    float c = 0.0f;
#pragma unroll
    for (int s = 0; s < NS; ++s)
      c += av[s] * mb[((size_t)src_idx(s) * B_DIM + b) * CTXD + col];
    __bf16 h = (__bf16)c;
    size_t idx = (size_t)b * CTXD + col;
    chi[idx] = h;
    clo[idx] = (__bf16)(c - (float)h);
  }
}

// ---------------------------------------------------------------------------
// Kernel D: attn[b][o] = [c|x|h1][b]·W_out[o] + b_out[o]  (1024x512, K=2560)
// Segment boundaries (1024, 1536) are multiples of 32 -> each K-step lives
// entirely in one source array; resolve the row base once per step.
// ---------------------------------------------------------------------------
__global__ __launch_bounds__(256) void k_out(const __bf16* __restrict__ chi,
                                             const __bf16* __restrict__ clo,
                                             const __bf16* __restrict__ xhi,
                                             const __bf16* __restrict__ xlo,
                                             const __bf16* __restrict__ hhi,
                                             const __bf16* __restrict__ hlo,
                                             const __bf16* __restrict__ Whi,
                                             const __bf16* __restrict__ Wlo,
                                             const float* __restrict__ bout,
                                             float* __restrict__ attn) {
  int lane = threadIdx.x & 31;
  int wave = threadIdx.x >> 5;
  int tile = blockIdx.x * 8 + wave;       // 0..2047
  int mbase = (tile >> 5) * 16;
  int nbase = (tile & 31) * 16;
  int b = mbase + (lane & 15);
  v8f acc = {};
  for (int k = 0; k < KCAT; k += 32) {
    const __bf16 *rhi, *rlo;
    int koff;
    if (k < CTXD)            { rhi = chi + (size_t)b * CTXD; rlo = clo + (size_t)b * CTXD; koff = k; }
    else if (k < CTXD + HID) { rhi = xhi + (size_t)b * HID;  rlo = xlo + (size_t)b * HID;  koff = k - CTXD; }
    else                     { rhi = hhi + (size_t)b * CTXD; rlo = hlo + (size_t)b * CTXD; koff = k - CTXD - HID; }
    v16bf ahi, alo, bhi, blo;
    load_a_frag_bf(rhi, rlo, koff, lane, ahi, alo);
    load_b_frag_bf(Whi, Wlo, KCAT, nbase, k, lane, bhi, blo);
    acc = wmma_bf16x3(ahi, alo, bhi, blo, acc);
  }
  int col = nbase + (lane & 15);
  float bias = bout[col];
#pragma unroll
  for (int r = 0; r < 8; ++r) {
    int row = mbase + r + 8 * (lane >> 4);
    attn[(size_t)row * HID + col] = acc[r] + bias;
  }
}

// ---------------------------------------------------------------------------
extern "C" void kernel_launch(void* const* d_in, const int* in_sizes, int n_in,
                              void* d_out, int out_size, void* d_ws, size_t ws_size,
                              hipStream_t stream) {
  const float* x    = (const float*)d_in[0];  // (1,1024,512)
  const float* h1   = (const float*)d_in[1];  // (1,1024,1024)
  const float* mb   = (const float*)d_in[2];  // (168,1024,1024)
  const float* Wctx = (const float*)d_in[3];  // (512,1024)
  const float* Wq   = (const float*)d_in[4];  // (512,512)
  const float* bq   = (const float*)d_in[5];  // (512,)
  const float* vv   = (const float*)d_in[6];  // (1,512)
  const float* Wout = (const float*)d_in[7];  // (512,2560)
  const float* bout = (const float*)d_in[8];  // (512,)
  (void)in_sizes; (void)n_in; (void)out_size; (void)ws_size;

  float* attn  = (float*)d_out;                        // B*512
  float* align = attn + (size_t)B_DIM * HID;           // B*13

  // ---- workspace layout (all regions >=16B aligned) ----------------------
  float* wq_buf = (float*)d_ws;                        // 524288 f32
  float* scores = wq_buf + (size_t)B_DIM * HID;        // 13312 f32 (pad 16384)
  __bf16* p = (__bf16*)(scores + 16384);
  __bf16* Wq_hi   = p;  p += (size_t)HID * HID;        // 262144
  __bf16* Wq_lo   = p;  p += (size_t)HID * HID;
  __bf16* Wctx_hi = p;  p += (size_t)HID * CTXD;       // 524288
  __bf16* Wctx_lo = p;  p += (size_t)HID * CTXD;
  __bf16* Wout_hi = p;  p += (size_t)HID * KCAT;       // 1310720
  __bf16* Wout_lo = p;  p += (size_t)HID * KCAT;
  __bf16* x_hi    = p;  p += (size_t)B_DIM * HID;      // 524288
  __bf16* x_lo    = p;  p += (size_t)B_DIM * HID;
  __bf16* h1_hi   = p;  p += (size_t)B_DIM * CTXD;     // 1048576
  __bf16* h1_lo   = p;  p += (size_t)B_DIM * CTXD;
  __bf16* c_hi    = p;  p += (size_t)B_DIM * CTXD;     // 1048576
  __bf16* c_lo    = p;                                  // total ws ~20.2 MB

  // ---- one-shot f32 -> bf16 hi/lo splits (memory bound, ~0.5us) ----------
  k_split<<<512, 256, 0, stream>>>(Wq,   Wq_hi,   Wq_lo,   HID * HID);
  k_split<<<512, 256, 0, stream>>>(Wctx, Wctx_hi, Wctx_lo, HID * CTXD);
  k_split<<<512, 256, 0, stream>>>(Wout, Wout_hi, Wout_lo, HID * KCAT);
  k_split<<<512, 256, 0, stream>>>(x,    x_hi,    x_lo,    B_DIM * HID);
  k_split<<<512, 256, 0, stream>>>(h1,   h1_hi,   h1_lo,   B_DIM * CTXD);

  // ---- pipeline ----------------------------------------------------------
  k_wq         <<<256, 256, 0, stream>>>(x_hi, x_lo, Wq_hi, Wq_lo, bq, wq_buf);
  k_uh_scores  <<<832, 256, 0, stream>>>(mb, Wctx_hi, Wctx_lo, wq_buf, vv, scores);
  k_softmax_ctx<<<128, 256, 0, stream>>>(mb, scores, align, c_hi, c_lo);
  k_out        <<<256, 256, 0, stream>>>(c_hi, c_lo, x_hi, x_lo, h1_hi, h1_lo,
                                         Wout_hi, Wout_lo, bout, attn);
}
